// GNN_21337397526641
// MI455X (gfx1250) — matrix operand order repaired and checked by reading, not compiled, not run on hardware
//
#include <hip/hip_runtime.h>
#include <hip/hip_bf16.h>

#define NNODES   100000
#define NEDGES   1600000
#define FDIM     128
#define NGRAPHS  512
#define NCLASSES 2
#define LDW      136   // padded LDS row stride in bf16 elems (128 + 8)

typedef __attribute__((ext_vector_type(16))) __bf16 v16bf;
typedef __attribute__((ext_vector_type(8)))  __bf16 v8bf;
typedef __attribute__((ext_vector_type(8)))  float  v8f;

// ---------------- utility kernels ----------------

__global__ void k_fill0(float* __restrict__ p, int n) {
    int i = blockIdx.x * blockDim.x + threadIdx.x;
    if (i < n) p[i] = 0.0f;
}

__global__ void k_deg(const int* __restrict__ dst, float* __restrict__ deg, int ne) {
    int e = blockIdx.x * blockDim.x + threadIdx.x;
    if (e < ne) atomicAdd(&deg[dst[e]], 1.0f);
}

__global__ void k_rsq(float* __restrict__ deg, int n) {
    int i = blockIdx.x * blockDim.x + threadIdx.x;
    if (i < n) deg[i] = rsqrtf(deg[i] + 1.0f);   // +1 = self loop; deg>=1 always
}

__global__ void k_cast_bf16(const float* __restrict__ x, __bf16* __restrict__ xb, int n) {
    int i = blockIdx.x * blockDim.x + threadIdx.x;
    if (i < n) xb[i] = (__bf16)x[i];
}

// ---------------- bf16 WMMA GEMM: H = A(bf16) @ W(bf16), A: nrows x 128, W: 128 x 128 ----------------

__global__ __launch_bounds__(256) void k_gemm_bf16(const __bf16* __restrict__ A,
                                                   const __bf16* __restrict__ W,
                                                   float* __restrict__ H, int nrows) {
    __shared__ __bf16 lw[128 * LDW];
    const int tid = threadIdx.x;

    // cooperative stage of W (128x128 bf16) into LDS, padded stride LDW
    for (int c = tid; c < 128 * 16; c += 256) {
        int k = c >> 4, seg = c & 15;
        *(v8bf*)&lw[k * LDW + seg * 8] = *(const v8bf*)&W[k * 128 + seg * 8];
    }
    __syncthreads();

    const int wave = tid >> 5;        // 0..7, each wave owns a 16-row strip
    const int lane = tid & 31;
    const int l15  = lane & 15;
    const int rowbase = blockIdx.x * 128 + wave * 16;

    int arow = rowbase + l15;
    if (arow >= nrows) arow = nrows - 1;          // clamp; stores are guarded
    // ISA 16-bit A layout: lanes 0-15 hold K 0-7 & 16-23, lanes 16-31 hold K 8-15 & 24-31
    const __bf16* ap = A + (size_t)arow * 128 + ((lane < 16) ? 0 : 8);

    v8f acc[8] = {};

    #pragma unroll
    for (int kk = 0; kk < 4; ++kk) {              // K steps of 32
        union { v16bf v; v8bf h[2]; } afrag;
        afrag.h[0] = *(const v8bf*)(ap + 32 * kk);
        afrag.h[1] = *(const v8bf*)(ap + 32 * kk + 16);
        // B layout: N = lane&15; lanes 0-15 hold K 0-7 & 16-23, lanes 16-31 hold K 8-15 & 24-31
        const __bf16* bp = &lw[(32 * kk + ((lane < 16) ? 0 : 8)) * LDW + l15];
        #pragma unroll
        for (int nt = 0; nt < 8; ++nt) {          // 8 N-tiles of 16
            union { v16bf v; v8bf h[2]; } bfrag;
            #pragma unroll
            for (int s = 0; s < 16; ++s)
                bfrag.v[s] = bp[((s < 8) ? s : s + 8) * LDW + nt * 16];
            acc[nt] = __builtin_amdgcn_wmma_f32_16x16x32_bf16(
                false, afrag.v, false, bfrag.v, (short)0, acc[nt], false, false);
        }
    }

    // C/D layout: VGPR r -> M = r (lanes 0-15) or r+8 (lanes 16-31); N = lane&15
    #pragma unroll
    for (int nt = 0; nt < 8; ++nt) {
        #pragma unroll
        for (int r = 0; r < 8; ++r) {
            int row = rowbase + ((lane < 16) ? r : r + 8);
            if (row < nrows) H[(size_t)row * 128 + nt * 16 + l15] = acc[nt][r];
        }
    }
}

// ---------------- message passing ----------------

// agg[i,:] = h[i,:] * dis[i]^2   (self-loop term; also serves as zero-init of agg)
__global__ void k_selfloop(const float* __restrict__ h, const float* __restrict__ dis,
                           float* __restrict__ agg, int n) {
    int i = blockIdx.x * blockDim.x + threadIdx.x;
    if (i < n * FDIM) {
        float d = dis[i >> 7];
        agg[i] = h[i] * d * d;
    }
}

// one wave per edge; lane handles 4 contiguous features
__global__ __launch_bounds__(256) void k_scatter(const int* __restrict__ src,
                                                 const int* __restrict__ dst,
                                                 const float* __restrict__ dis,
                                                 const float* __restrict__ h,
                                                 float* __restrict__ agg, int ne) {
    int gidx = blockIdx.x * blockDim.x + threadIdx.x;
    int e = gidx >> 5;
    if (e >= ne) return;
    int lane = gidx & 31;
    int s = src[e], d = dst[e];
    float w = dis[s] * dis[d];
    float4 v = ((const float4*)(h + (size_t)s * FDIM))[lane];
    float* ap = agg + (size_t)d * FDIM + lane * 4;
    atomicAdd(ap + 0, v.x * w);
    atomicAdd(ap + 1, v.y * w);
    atomicAdd(ap + 2, v.z * w);
    atomicAdd(ap + 3, v.w * w);
}

__global__ void k_bias_relu_bf16(const float* __restrict__ agg, const float* __restrict__ b,
                                 __bf16* __restrict__ xb, int n) {
    int i = blockIdx.x * blockDim.x + threadIdx.x;
    if (i < n * FDIM) {
        float v = agg[i] + b[i & 127];
        xb[i] = (__bf16)(v > 0.0f ? v : 0.0f);
    }
}

__global__ void k_bias_relu_f32(float* __restrict__ agg, const float* __restrict__ b, int n) {
    int i = blockIdx.x * blockDim.x + threadIdx.x;
    if (i < n * FDIM) {
        float v = agg[i] + b[i & 127];
        agg[i] = v > 0.0f ? v : 0.0f;
    }
}

// ---------------- pooling + head ----------------

__global__ void k_pool(const int* __restrict__ batch, const float* __restrict__ x,
                       float* __restrict__ sums, float* __restrict__ counts, int n) {
    int gidx = blockIdx.x * blockDim.x + threadIdx.x;
    int node = gidx >> 5;
    if (node >= n) return;
    int lane = gidx & 31;
    int g = batch[node];
    float4 v = ((const float4*)(x + (size_t)node * FDIM))[lane];
    float* sp = sums + g * FDIM + lane * 4;
    atomicAdd(sp + 0, v.x);
    atomicAdd(sp + 1, v.y);
    atomicAdd(sp + 2, v.z);
    atomicAdd(sp + 3, v.w);
    if (lane == 0) atomicAdd(&counts[g], 1.0f);
}

__global__ void k_head(const float* __restrict__ sums, const float* __restrict__ counts,
                       const float* __restrict__ Wlin, const float* __restrict__ blin,
                       float* __restrict__ out) {
    int idx = blockIdx.x * blockDim.x + threadIdx.x;
    if (idx >= NGRAPHS * NCLASSES) return;
    int g = idx / NCLASSES, c = idx % NCLASSES;
    float inv = 1.0f / fmaxf(counts[g], 1.0f);
    float acc = 0.0f;
    for (int k = 0; k < FDIM; ++k)
        acc += sums[g * FDIM + k] * inv * Wlin[k * NCLASSES + c];
    out[idx] = acc + blin[c];
}

// ---------------- launch ----------------

extern "C" void kernel_launch(void* const* d_in, const int* in_sizes, int n_in,
                              void* d_out, int out_size, void* d_ws, size_t ws_size,
                              hipStream_t stream) {
    const float* x    = (const float*)d_in[0];
    const int*   ei   = (const int*)d_in[1];     // (2, NEDGES): src then dst
    const int*   batch= (const int*)d_in[2];
    const float* W1   = (const float*)d_in[3];
    const float* b1   = (const float*)d_in[4];
    const float* W2   = (const float*)d_in[5];
    const float* b2   = (const float*)d_in[6];
    const float* Wlin = (const float*)d_in[7];
    const float* blin = (const float*)d_in[8];
    float* out = (float*)d_out;

    const int* src = ei;
    const int* dst = ei + NEDGES;

    // workspace bump allocator (256B aligned)
    char* wsp = (char*)d_ws;
    auto alloc = [&](size_t bytes) -> void* {
        void* p = wsp;
        wsp += (bytes + 255) & ~(size_t)255;
        return p;
    };
    float*  dis  = (float*)alloc((size_t)NNODES * 4);
    __bf16* xb   = (__bf16*)alloc((size_t)NNODES * FDIM * 2);
    float*  h    = (float*)alloc((size_t)NNODES * FDIM * 4);
    float*  agg  = (float*)alloc((size_t)NNODES * FDIM * 4);
    __bf16* w1b  = (__bf16*)alloc(FDIM * FDIM * 2);
    __bf16* w2b  = (__bf16*)alloc(FDIM * FDIM * 2);
    float*  sums = (float*)alloc(NGRAPHS * FDIM * 4);
    float*  cnts = (float*)alloc(NGRAPHS * 4);

    const int T = 256;
    const int NF = NNODES * FDIM;                         // 12.8M
    const int gemmBlocks = (NNODES + 127) / 128;          // 782
    const int scatBlocks = (NEDGES * 32 + T - 1) / T;     // one wave per edge

    // degree -> dis
    k_fill0<<<(NNODES + T - 1) / T, T, 0, stream>>>(dis, NNODES);
    k_deg<<<(NEDGES + T - 1) / T, T, 0, stream>>>(dst, dis, NEDGES);
    k_rsq<<<(NNODES + T - 1) / T, T, 0, stream>>>(dis, NNODES);

    // casts
    k_cast_bf16<<<(NF + T - 1) / T, T, 0, stream>>>(x, xb, NF);
    k_cast_bf16<<<(FDIM * FDIM + T - 1) / T, T, 0, stream>>>(W1, w1b, FDIM * FDIM);
    k_cast_bf16<<<(FDIM * FDIM + T - 1) / T, T, 0, stream>>>(W2, w2b, FDIM * FDIM);

    // layer 1: transform -> propagate -> bias+relu (to bf16 for next GEMM)
    k_gemm_bf16<<<gemmBlocks, T, 0, stream>>>(xb, w1b, h, NNODES);
    k_selfloop<<<(NF + T - 1) / T, T, 0, stream>>>(h, dis, agg, NNODES);
    k_scatter<<<scatBlocks, T, 0, stream>>>(src, dst, dis, h, agg, NEDGES);
    k_bias_relu_bf16<<<(NF + T - 1) / T, T, 0, stream>>>(agg, b1, xb, NNODES);

    // layer 2
    k_gemm_bf16<<<gemmBlocks, T, 0, stream>>>(xb, w2b, h, NNODES);
    k_selfloop<<<(NF + T - 1) / T, T, 0, stream>>>(h, dis, agg, NNODES);
    k_scatter<<<scatBlocks, T, 0, stream>>>(src, dst, dis, h, agg, NEDGES);
    k_bias_relu_f32<<<(NF + T - 1) / T, T, 0, stream>>>(agg, b2, NNODES);

    // mean pool + linear head
    k_fill0<<<(NGRAPHS * FDIM + T - 1) / T, T, 0, stream>>>(sums, NGRAPHS * FDIM);
    k_fill0<<<(NGRAPHS + T - 1) / T, T, 0, stream>>>(cnts, NGRAPHS);
    k_pool<<<(NNODES * 32 + T - 1) / T, T, 0, stream>>>(batch, agg, sums, cnts, NNODES);
    k_head<<<(NGRAPHS * NCLASSES + T - 1) / T, T, 0, stream>>>(sums, cnts, Wlin, blin, out);
}
